// Model_63247688401137
// MI455X (gfx1250) — matrix operand order repaired and checked by reading, not compiled, not run on hardware
//
#include <hip/hip_runtime.h>
#include <hip/hip_bf16.h>
#include <math.h>

#define B_      2
#define S_      2048
#define D_      512
#define H_      8
#define DPH_    64
#define DH_     2048
#define NLAYER_ 2
#define NOUT_   32000
#define ROWS_   (B_ * S_)   /* 4096 */

#define KCH_    256          /* K super-chunk staged by TDM           */
#define BROW_   264          /* LDS row stride in u16: 512B data + 16B pad */

typedef unsigned short u16;
typedef __attribute__((ext_vector_type(16))) __bf16 v16bf;
typedef __attribute__((ext_vector_type(8)))  float  v8f;
typedef __attribute__((ext_vector_type(4)))  unsigned int u32x4;
typedef __attribute__((ext_vector_type(8)))  int i32x8;
typedef __attribute__((ext_vector_type(4)))  int i32x4;

union FRAG { v16bf v; u16 u[16]; uint4 q[2]; };
union ACC8 { v8f v; float e[8]; };

__device__ __forceinline__ u16 f2bfbits(float f) {
  unsigned int u = __float_as_uint(f);
  u += 0x7FFFu + ((u >> 16) & 1u);   // round-to-nearest-even
  return (u16)(u >> 16);
}

// ---------------------------------------------------------------------------
// f32 -> bf16 bulk conversion (weights); n % 4 == 0
// ---------------------------------------------------------------------------
__global__ __launch_bounds__(256)
void cvt_kernel(const float* __restrict__ src, u16* __restrict__ dst, long long n) {
  long long i = ((long long)blockIdx.x * blockDim.x + threadIdx.x) * 4;
  if (i >= n) return;
  float4 a = *(const float4*)(src + i);
  unsigned int lo = (unsigned int)f2bfbits(a.x) | ((unsigned int)f2bfbits(a.y) << 16);
  unsigned int hi = (unsigned int)f2bfbits(a.z) | ((unsigned int)f2bfbits(a.w) << 16);
  uint2 o; o.x = lo; o.y = hi;
  *(uint2*)(dst + i) = o;
}

// ---------------------------------------------------------------------------
// Embedding gather with dual f32 + bf16 output
// ---------------------------------------------------------------------------
__global__ __launch_bounds__(256)
void embed_kernel(const int* __restrict__ x, const float* __restrict__ Wt,
                  float* __restrict__ e, u16* __restrict__ ebf) {
  long long gid = (long long)blockIdx.x * blockDim.x + threadIdx.x;
  if (gid >= (long long)ROWS_ * D_) return;
  int row = (int)(gid >> 9);
  int d   = (int)(gid & 511);
  int tok = x[row];
  float v = Wt[(long long)tok * D_ + d];
  e[gid] = v;
  ebf[gid] = f2bfbits(v);
}

// ---------------------------------------------------------------------------
// GEMM: C[M,N] = A[M,K] @ W[N,K]^T (+bias)(relu). A, W in bf16.
// 128 threads = 4 waves; block tile 128x64; wave tile 32x64 (8 WMMA tiles).
// W panel (64 x KCH_) staged into LDS by the Tensor Data Mover (wave 0 issues
// the D#, waits TENSORcnt), with D# padding (16B per 512B row) for bank-
// conflict-free ds_load_b128 fragment reads shared by all 4 waves.
// Dual output: f32 (Cf) and/or bf16 (Cb). M%128==0, N%64==0, K%KCH_==0.
// ---------------------------------------------------------------------------
__global__ __launch_bounds__(128)
void gemm_bf16(const u16* __restrict__ A, const u16* __restrict__ W,
               const float* __restrict__ bias, float* __restrict__ Cf,
               u16* __restrict__ Cb, int N, int K, int relu) {
  __shared__ u16 bt[64 * BROW_];                 // 33,792 B

  const int lane = threadIdx.x & 31;
  const int warp = threadIdx.x >> 5;
  const int half = lane >> 4;
  const int hn   = lane & 15;
  const int cb = blockIdx.x * 64;
  const int rb = blockIdx.y * 128 + warp * 32;

  ACC8 acc[2][4];
#pragma unroll
  for (int rt = 0; rt < 2; ++rt)
#pragma unroll
    for (int t = 0; t < 4; ++t)
#pragma unroll
      for (int r = 0; r < 8; ++r) acc[rt][t].e[r] = 0.f;

  const long long arow0 = (long long)(rb + hn) * K;
  const long long arow1 = (long long)(rb + 16 + hn) * K;

  for (int kc = 0; kc < K; kc += KCH_) {
    // ---- TDM: DMA 64 x KCH_ bf16 panel of W into LDS (wave 0 only) ----
    if (threadIdx.x < 32) {
      unsigned long long ga =
          (unsigned long long)(size_t)(W + (size_t)cb * K + kc);
      unsigned ldsa = (unsigned)(size_t)(void*)bt;   // addr[31:0] = LDS offset
      u32x4 g0;
      g0[0] = 1u;                                    // count=1 (valid D#)
      g0[1] = ldsa;                                  // lds_addr (bytes)
      g0[2] = (unsigned)ga;                          // global_addr[31:0]
      g0[3] = (unsigned)((ga >> 32) & 0x1FFFFFFu)    // global_addr[56:32]
              | (2u << 30);                          // type = 2 ("image")
      i32x8 g1;
      g1[0] = (int)((1u << 16)                       // data_size = 2 bytes
                    | (1u << 20)                     // pad_enable
                    | (6u << 22)                     // pad_interval: 128 DW = 512B
                    | (3u << 25));                   // pad_amount: 4 DW = 16B
      g1[1] = (int)(((unsigned)K & 0xFFFFu) << 16);  // tensor_dim0[15:0]
      g1[2] = (int)(((unsigned)K >> 16) & 0xFFFFu);  // tensor_dim0[31:16] | dim1 lo (0)
      g1[3] = (int)(((1u << 20) >> 16)               // tensor_dim1[31:16] (dim1 = 2^20)
                    | ((unsigned)KCH_ << 16));       // tile_dim0 = KCH_
      g1[4] = 64;                                    // tile_dim1 = 64, tile_dim2 = 0
      g1[5] = (int)(unsigned)K;                      // tensor_dim0_stride[31:0]
      g1[6] = 0;                                     // stride0 hi | stride1 lo
      g1[7] = 0;                                     // stride1 hi
      i32x4 gz; gz[0] = 0; gz[1] = 0; gz[2] = 0; gz[3] = 0;
      i32x8 gz8;
      gz8[0] = 0; gz8[1] = 0; gz8[2] = 0; gz8[3] = 0;
      gz8[4] = 0; gz8[5] = 0; gz8[6] = 0; gz8[7] = 0;
      __builtin_amdgcn_tensor_load_to_lds(g0, g1, gz, gz, gz8, 0);
      __builtin_amdgcn_s_wait_tensorcnt(0);
    }
    __syncthreads();

    for (int k0 = 0; k0 < KCH_; k0 += 32) {
      FRAG a0, a1;
      a0.q[0] = *(const uint4*)(A + arow0 + kc + k0 + 8 * half);
      a0.q[1] = *(const uint4*)(A + arow0 + kc + k0 + 16 + 8 * half);
      a1.q[0] = *(const uint4*)(A + arow1 + kc + k0 + 8 * half);
      a1.q[1] = *(const uint4*)(A + arow1 + kc + k0 + 16 + 8 * half);
#pragma unroll
      for (int t = 0; t < 4; ++t) {
        FRAG wf;
        const u16* brow = bt + (t * 16 + hn) * BROW_ + k0 + 8 * half;
        wf.q[0] = *(const uint4*)(brow);
        wf.q[1] = *(const uint4*)(brow + 16);
        acc[0][t].v = __builtin_amdgcn_wmma_f32_16x16x32_bf16(
            false, a0.v, false, wf.v, (short)0, acc[0][t].v, false, false);
        acc[1][t].v = __builtin_amdgcn_wmma_f32_16x16x32_bf16(
            false, a1.v, false, wf.v, (short)0, acc[1][t].v, false, false);
      }
    }
    __syncthreads();   // protect LDS before next TDM overwrite
  }

#pragma unroll
  for (int rt = 0; rt < 2; ++rt)
#pragma unroll
    for (int t = 0; t < 4; ++t) {
      int col = cb + t * 16 + hn;
      float bv = bias ? bias[col] : 0.f;
#pragma unroll
      for (int r = 0; r < 8; ++r) {
        int row = rb + rt * 16 + r + 8 * half;
        float v = acc[rt][t].e[r] + bv;
        if (relu) v = v > 0.f ? v : 0.f;
        long long idx = (long long)row * N + col;
        if (Cf) Cf[idx] = v;
        if (Cb) Cb[idx] = f2bfbits(v);
      }
    }
}

// ---------------------------------------------------------------------------
// Flash attention: per (b,h), softmax((Q K^T + mask)/sqrt(64)) @ V.
// Q/K/V in bf16 ([B*S, D], head h at cols [h*64, h*64+64)); O in f32.
// One wave = 16 query rows; online softmax over 32-key chunks.
// ---------------------------------------------------------------------------
__global__ __launch_bounds__(128)
void flash_attn(const u16* __restrict__ Q, const u16* __restrict__ Km,
                const u16* __restrict__ V, const float* __restrict__ mask,
                float* __restrict__ O) {
  __shared__ u16 plds[4][16][32];

  const int lane = threadIdx.x & 31;
  const int warp = threadIdx.x >> 5;
  const int half = lane >> 4;
  const int hn   = lane & 15;
  const int wid = blockIdx.x * 4 + warp;       // 0..2047
  const int qt = wid & 127;
  const int h  = (wid >> 7) & 7;
  const int b  = wid >> 10;
  const int qbase = qt * 16;

  FRAG qa[2];
  {
    const long long rowQ = ((long long)(b * S_ + qbase + hn)) * D_ + h * DPH_;
#pragma unroll
    for (int kk = 0; kk < 2; ++kk) {
      const u16* base = Q + rowQ + kk * 32;
      qa[kk].q[0] = *(const uint4*)(base + 8 * half);
      qa[kk].q[1] = *(const uint4*)(base + 16 + 8 * half);
    }
  }

  ACC8 o_acc[4];
  float mrow[8], lrow[8];
#pragma unroll
  for (int t = 0; t < 4; ++t)
#pragma unroll
    for (int r = 0; r < 8; ++r) o_acc[t].e[r] = 0.f;
#pragma unroll
  for (int r = 0; r < 8; ++r) { mrow[r] = -1e30f; lrow[r] = 0.f; }

  for (int jb = 0; jb < S_; jb += 32) {
    ACC8 s0, s1;
#pragma unroll
    for (int r = 0; r < 8; ++r) { s0.e[r] = 0.f; s1.e[r] = 0.f; }
#pragma unroll
    for (int kk = 0; kk < 2; ++kk) {
      FRAG kb0, kb1;
      const long long kr0 =
          ((long long)(b * S_ + jb + hn)) * D_ + h * DPH_ + kk * 32 + 8 * half;
      const long long kr1 =
          ((long long)(b * S_ + jb + 16 + hn)) * D_ + h * DPH_ + kk * 32 + 8 * half;
      kb0.q[0] = *(const uint4*)(Km + kr0);
      kb0.q[1] = *(const uint4*)(Km + kr0 + 16);
      kb1.q[0] = *(const uint4*)(Km + kr1);
      kb1.q[1] = *(const uint4*)(Km + kr1 + 16);
      s0.v = __builtin_amdgcn_wmma_f32_16x16x32_bf16(
          false, qa[kk].v, false, kb0.v, (short)0, s0.v, false, false);
      s1.v = __builtin_amdgcn_wmma_f32_16x16x32_bf16(
          false, qa[kk].v, false, kb1.v, (short)0, s1.v, false, false);
    }

    float scale_r[8];
#pragma unroll
    for (int r = 0; r < 8; ++r) {
      int m_c = r + 8 * half;
      long long moff = (long long)(qbase + m_c) * S_;
      float v0 = (s0.e[r] + mask[moff + jb + hn]) * 0.125f;
      float v1 = (s1.e[r] + mask[moff + jb + 16 + hn]) * 0.125f;
      float mx = fmaxf(v0, v1);
      mx = fmaxf(mx, __shfl_xor(mx, 1, 32));
      mx = fmaxf(mx, __shfl_xor(mx, 2, 32));
      mx = fmaxf(mx, __shfl_xor(mx, 4, 32));
      mx = fmaxf(mx, __shfl_xor(mx, 8, 32));
      float mnew = fmaxf(mrow[r], mx);
      float sc = __expf(mrow[r] - mnew);
      float p0 = __expf(v0 - mnew);
      float p1 = __expf(v1 - mnew);
      float sm = p0 + p1;
      sm += __shfl_xor(sm, 1, 32);
      sm += __shfl_xor(sm, 2, 32);
      sm += __shfl_xor(sm, 4, 32);
      sm += __shfl_xor(sm, 8, 32);
      lrow[r] = lrow[r] * sc + sm;
      mrow[r] = mnew;
      scale_r[r] = sc;
      plds[warp][m_c][hn]      = f2bfbits(p0);
      plds[warp][m_c][16 + hn] = f2bfbits(p1);
    }
#pragma unroll
    for (int t = 0; t < 4; ++t)
#pragma unroll
      for (int r = 0; r < 8; ++r) o_acc[t].e[r] *= scale_r[r];

    __syncthreads();
    FRAG pa;
#pragma unroll
    for (int e = 0; e < 16; ++e) {
      int k = (e < 8) ? (e + 8 * half) : (e + 8 + 8 * half);
      pa.u[e] = plds[warp][hn][k];
    }
    __syncthreads();

#pragma unroll
    for (int t = 0; t < 4; ++t) {
      FRAG vb;
#pragma unroll
      for (int e = 0; e < 16; ++e) {
        int krow = (e < 8) ? (e + 8 * half) : (e + 8 + 8 * half);
        vb.u[e] = V[((long long)(b * S_ + jb + krow)) * D_ + h * DPH_ + t * 16 + hn];
      }
      o_acc[t].v = __builtin_amdgcn_wmma_f32_16x16x32_bf16(
          false, pa.v, false, vb.v, (short)0, o_acc[t].v, false, false);
    }
  }

#pragma unroll
  for (int t = 0; t < 4; ++t)
#pragma unroll
    for (int r = 0; r < 8; ++r) {
      int m_c = r + 8 * half;
      O[((long long)(b * S_ + qbase + m_c)) * D_ + h * DPH_ + t * 16 + hn] =
          o_acc[t].e[r] / lrow[r];
    }
}

// ---------------------------------------------------------------------------
// LayerNorm over D=512, fused residual, dual f32 + optional bf16 output.
// One wave per row; 8 rows per 256-thread block.
// ---------------------------------------------------------------------------
__global__ __launch_bounds__(256)
void ln_kernel(const float* __restrict__ X, const float* __restrict__ Rres,
               const float* __restrict__ g, const float* __restrict__ bta,
               float* __restrict__ Out, u16* __restrict__ OutBf, int rows) {
  int lane = threadIdx.x & 31;
  int wv   = threadIdx.x >> 5;
  int row  = blockIdx.x * 8 + wv;
  if (row >= rows) return;
  const float* xp = X + (long long)row * D_;
  const float* rp = Rres + (long long)row * D_;

  float vals[16];
  float s = 0.f;
#pragma unroll
  for (int i = 0; i < 4; ++i) {
    float4 a = ((const float4*)(xp + lane * 16))[i];
    float4 b = ((const float4*)(rp + lane * 16))[i];
    a.x += b.x; a.y += b.y; a.z += b.z; a.w += b.w;
    vals[4 * i + 0] = a.x; vals[4 * i + 1] = a.y;
    vals[4 * i + 2] = a.z; vals[4 * i + 3] = a.w;
    s += a.x + a.y + a.z + a.w;
  }
  s += __shfl_xor(s, 1, 32);  s += __shfl_xor(s, 2, 32);
  s += __shfl_xor(s, 4, 32);  s += __shfl_xor(s, 8, 32);
  s += __shfl_xor(s, 16, 32);
  float mu = s * (1.f / D_);
  float vs = 0.f;
#pragma unroll
  for (int i = 0; i < 16; ++i) { float d = vals[i] - mu; vs += d * d; }
  vs += __shfl_xor(vs, 1, 32);  vs += __shfl_xor(vs, 2, 32);
  vs += __shfl_xor(vs, 4, 32);  vs += __shfl_xor(vs, 8, 32);
  vs += __shfl_xor(vs, 16, 32);
  float inv = rsqrtf(vs * (1.f / D_) + 1e-5f);
#pragma unroll
  for (int i = 0; i < 16; ++i) {
    int d = lane * 16 + i;
    float v = (vals[i] - mu) * inv * g[d] + bta[d];
    long long idx = (long long)row * D_ + d;
    Out[idx] = v;
    if (OutBf) OutBf[idx] = f2bfbits(v);
  }
}

// ---------------------------------------------------------------------------
extern "C" void kernel_launch(void* const* d_in, const int* in_sizes, int n_in,
                              void* d_out, int out_size, void* d_ws, size_t ws_size,
                              hipStream_t stream) {
  (void)in_sizes; (void)n_in; (void)out_size; (void)ws_size;
  const int*   x      = (const int*)d_in[0];
  const float* mask   = (const float*)d_in[1];
  const float* embedW = (const float*)d_in[2];
  const float* Wq     = (const float*)d_in[3];
  const float* Wk     = (const float*)d_in[4];
  const float* Wv     = (const float*)d_in[5];
  const float* w1W    = (const float*)d_in[6];
  const float* w1b    = (const float*)d_in[7];
  const float* w2W    = (const float*)d_in[8];
  const float* w2b    = (const float*)d_in[9];
  const float* lng    = (const float*)d_in[10];
  const float* lnb    = (const float*)d_in[11];
  const float* ln2g   = (const float*)d_in[12];
  const float* ln2b   = (const float*)d_in[13];
  const float* fW     = (const float*)d_in[14];
  const float* fb     = (const float*)d_in[15];

  float* out = (float*)d_out;
  const size_t LOG   = (size_t)ROWS_ * NOUT_;
  const size_t HIDSZ = (size_t)ROWS_ * DH_;

  // ---- workspace layout ----
  const size_t U = (size_t)ROWS_ * D_;
  float* E  = (float*)d_ws;
  float* Ob = E  + U;
  float* O1 = Ob + U;
  float* T2 = O1 + U;
  u16* Ebf   = (u16*)(T2 + U);
  u16* O1bf  = Ebf  + U;
  u16* Qbf   = O1bf + U;
  u16* Kbf   = Qbf  + U;
  u16* Vbf   = Kbf  + U;
  u16* HidBf = Vbf  + U;
  u16* WqB = HidBf + (size_t)ROWS_ * DH_;
  u16* WkB = WqB + (size_t)NLAYER_ * D_ * D_;
  u16* WvB = WkB + (size_t)NLAYER_ * D_ * D_;
  u16* W1B = WvB + (size_t)NLAYER_ * D_ * D_;
  u16* W2B = W1B + (size_t)NLAYER_ * DH_ * D_;
  u16* FWB = W2B + (size_t)NLAYER_ * D_ * DH_;

  {
    long long nqkv = (long long)NLAYER_ * D_ * D_;
    long long nffn = (long long)NLAYER_ * DH_ * D_;
    long long nfin = (long long)NOUT_ * D_;
    cvt_kernel<<<(unsigned)((nqkv / 4 + 255) / 256), 256, 0, stream>>>(Wq, WqB, nqkv);
    cvt_kernel<<<(unsigned)((nqkv / 4 + 255) / 256), 256, 0, stream>>>(Wk, WkB, nqkv);
    cvt_kernel<<<(unsigned)((nqkv / 4 + 255) / 256), 256, 0, stream>>>(Wv, WvB, nqkv);
    cvt_kernel<<<(unsigned)((nffn / 4 + 255) / 256), 256, 0, stream>>>(w1W, W1B, nffn);
    cvt_kernel<<<(unsigned)((nffn / 4 + 255) / 256), 256, 0, stream>>>(w2W, W2B, nffn);
    cvt_kernel<<<(unsigned)((nfin / 4 + 255) / 256), 256, 0, stream>>>(fW, FWB, nfin);
  }

  embed_kernel<<<(ROWS_ * D_ + 255) / 256, 256, 0, stream>>>(x, embedW, E, Ebf);

  for (int l = 0; l < NLAYER_; ++l) {
    dim3 gqkv(D_ / 64, ROWS_ / 128);
    gemm_bf16<<<gqkv, 128, 0, stream>>>(Ebf, WqB + (size_t)l * D_ * D_, nullptr,
                                        nullptr, Qbf, D_, D_, 0);
    gemm_bf16<<<gqkv, 128, 0, stream>>>(Ebf, WkB + (size_t)l * D_ * D_, nullptr,
                                        nullptr, Kbf, D_, D_, 0);
    gemm_bf16<<<gqkv, 128, 0, stream>>>(Ebf, WvB + (size_t)l * D_ * D_, nullptr,
                                        nullptr, Vbf, D_, D_, 0);

    flash_attn<<<(B_ * H_ * (S_ / 16)) / 4, 128, 0, stream>>>(Qbf, Kbf, Vbf, mask, Ob);

    ln_kernel<<<ROWS_ / 8, 256, 0, stream>>>(Ob, E, lng + l * D_, lnb + l * D_,
                                             O1, O1bf, ROWS_);

    float* hid = out + LOG + (size_t)l * HIDSZ;
    dim3 g1(DH_ / 64, ROWS_ / 128);
    gemm_bf16<<<g1, 128, 0, stream>>>(O1bf, W1B + (size_t)l * DH_ * D_,
                                      w1b + (size_t)l * DH_, hid, HidBf,
                                      DH_, D_, 1);
    dim3 g2(D_ / 64, ROWS_ / 128);
    gemm_bf16<<<g2, 128, 0, stream>>>(HidBf, W2B + (size_t)l * D_ * DH_,
                                      w2b + (size_t)l * D_, T2, nullptr,
                                      D_, DH_, 0);
    ln_kernel<<<ROWS_ / 8, 256, 0, stream>>>(T2, O1, ln2g + l * D_,
                                             ln2b + l * D_, E, Ebf, ROWS_);
  }

  dim3 gf(NOUT_ / 64, ROWS_ / 128);
  gemm_bf16<<<gf, 128, 0, stream>>>(Ebf, FWB, fb, out, nullptr, NOUT_, D_, 0);
}
